// PIBNet_89163521065279
// MI455X (gfx1250) — compile-verified, matrix operand order checked by reading, not run on hardware
//
#include <hip/hip_runtime.h>

// ---------------------------------------------------------------------------
// MeshGraphNet-style multi-level GNN forward for MI455X (gfx1250).
// All GEMMs run on v_wmma_f32_16x16x32_bf16 (wave32 WMMA, f32 accumulate).
// f32 master activations in memory; bf16 only as WMMA operands.
// M-tile = 32 rows/workgroup: each wave reuses every B (weight) fragment
// across two A tiles -> half the weight traffic per FLOP.
// ---------------------------------------------------------------------------

typedef unsigned int  u32;
typedef unsigned short u16;
typedef __attribute__((ext_vector_type(16))) __bf16 v16bf;
typedef __attribute__((ext_vector_type(8)))  float  v8f;

union FragU { u32 u[8]; v16bf v; };

__device__ __forceinline__ u16 f2bf(float x) {
    u32 u = __float_as_uint(x);
    u32 r = u + 0x7FFFu + ((u >> 16) & 1u);   // round-to-nearest-even
    return (u16)(r >> 16);
}

// A-fragment (16x32 bf16) from LDS, row-major bf16 rows of length K.
// Per ISA 7.12.2: lane L holds row M=L&15; VGPR i holds K = (i>=4?16:0) + 8*(L>>4) + 2*(i&3) [+1].
__device__ __forceinline__ v16bf load_a(const u32* X, int strideU, int kk, int lane) {
    FragU f;
    const int row = lane & 15;
    const int g8  = (lane >> 4) * 8;
#pragma unroll
    for (int i = 0; i < 8; ++i) {
        int k0 = kk + ((i & 4) << 2) + g8 + ((i & 3) << 1);
        f.u[i] = X[row * strideU + (k0 >> 1)];
    }
    return f.v;
}

// B-fragment (32x16 bf16): lane holds column N=L&15; same K packing as A.
// Weights pre-transposed to [col][k] bf16 so each pair is one aligned u32 load.
__device__ __forceinline__ v16bf load_b(const u16* wcol, int kk, int lane) {
    FragU f;
    const int g8 = (lane >> 4) * 8;
#pragma unroll
    for (int i = 0; i < 8; ++i) {
        int k0 = kk + ((i & 4) << 2) + g8 + ((i & 3) << 1);
        f.u[i] = *(const u32*)(wcol + k0);
    }
    return f.v;
}

// 32xK @ Kx16 -> two 16x16 f32 tiles; B fragment shared across both A tiles.
template<int K>
__device__ __forceinline__ void gemm_tile2(const u32* X, const u16* Wt, int colBase, int lane,
                                           v8f& c0, v8f& c1) {
    const u16* wcol = Wt + (size_t)(colBase + (lane & 15)) * K;
#pragma unroll
    for (int kk = 0; kk < K; kk += 32) {
        v16bf b  = load_b(wcol, kk, lane);
        v16bf a0 = load_a(X,                K / 2, kk, lane);   // rows 0-15
        v16bf a1 = load_a(X + 16 * (K / 2), K / 2, kk, lane);   // rows 16-31
        c0 = __builtin_amdgcn_wmma_f32_16x16x32_bf16(false, a0, false, b, (short)0, c0, false, false);
        c1 = __builtin_amdgcn_wmma_f32_16x16x32_bf16(false, a1, false, b, (short)0, c1, false, false);
    }
}

// Fused Edge/Node block: gather(+concat) -> 3-layer MLP (WMMA) -> LayerNorm
// -> residual -> [store] + [atomic scatter-sum into agg].
// Workgroup = 256 threads (8 waves) handles 32 rows; wave w owns cols [16w,16w+16).
template<int KIN, bool EDGE>
__global__ __launch_bounds__(256) void mgn_block_kernel(
    const float* __restrict__ featA,   // EDGE: efeat[E][128]      NODE: agg[N][128]
    const float* __restrict__ nodes,   // EDGE: gather table       NODE: nf_in[N][128]
    const int*   __restrict__ srcIdx,  // EDGE only
    const int*   __restrict__ dstIdx,  // EDGE only
    const float* __restrict__ resid,   // EDGE: efeat              NODE: nf_out
    const u16* __restrict__ Wt1, const float* __restrict__ b1,
    const u16* __restrict__ Wt2, const float* __restrict__ b2,
    const u16* __restrict__ Wt3, const float* __restrict__ b3,
    const float* __restrict__ gamma, const float* __restrict__ beta,
    float* __restrict__ outBuf,        // may be null
    float* __restrict__ aggBuf,        // EDGE only (scatter-sum target)
    int nrows)
{
    __shared__ u32   Xs [32 * KIN / 2];
    __shared__ u32   H1s[32 * 128 / 2];
    __shared__ u32   H2s[32 * 128 / 2];
    __shared__ float Os [32 * 128];
    __shared__ float red[32][8][2];
    __shared__ float stats[32][2];

    const int tid     = threadIdx.x;
    const int rowBase = blockIdx.x * 32;

    // ---- stage gathered/concatenated input rows as bf16 ----
    u16* x16 = (u16*)Xs;
    for (int idx = tid; idx < 32 * KIN; idx += 256) {
        int r = idx / KIN;
        int c = idx - r * KIN;
        int e = rowBase + r; if (e >= nrows) e = nrows - 1;
        float v;
        if (EDGE) {
            if (c < 128)      v = featA[(size_t)e * 128 + c];
            else if (c < 256) v = nodes[(size_t)srcIdx[e] * 128 + (c - 128)];
            else              v = nodes[(size_t)dstIdx[e] * 128 + (c - 256)];
        } else {
            v = (c < 128) ? featA[(size_t)e * 128 + c]
                          : nodes[(size_t)e * 128 + (c - 128)];
        }
        x16[idx] = f2bf(v);
    }
    __syncthreads();

    const int lane    = tid & 31;
    const int wave    = tid >> 5;
    const int colBase = wave * 16;
    const int col     = colBase + (lane & 15);
    const int g8row   = (lane >> 4) * 8;

    // ---- layer 1: KIN -> 128, ReLU ----
    {
        v8f c0 = {0.f,0.f,0.f,0.f,0.f,0.f,0.f,0.f};
        v8f c1 = {0.f,0.f,0.f,0.f,0.f,0.f,0.f,0.f};
        gemm_tile2<KIN>(Xs, Wt1, colBase, lane, c0, c1);
        float bb = b1[col];
        u16* h16 = (u16*)H1s;
#pragma unroll
        for (int r = 0; r < 8; ++r) {
            float v0 = c0[r] + bb; v0 = v0 > 0.f ? v0 : 0.f;
            float v1 = c1[r] + bb; v1 = v1 > 0.f ? v1 : 0.f;
            h16[(r + g8row) * 128 + col]      = f2bf(v0);
            h16[(r + g8row + 16) * 128 + col] = f2bf(v1);
        }
    }
    __syncthreads();

    // ---- layer 2: 128 -> 128, ReLU ----
    {
        v8f c0 = {0.f,0.f,0.f,0.f,0.f,0.f,0.f,0.f};
        v8f c1 = {0.f,0.f,0.f,0.f,0.f,0.f,0.f,0.f};
        gemm_tile2<128>(H1s, Wt2, colBase, lane, c0, c1);
        float bb = b2[col];
        u16* h16 = (u16*)H2s;
#pragma unroll
        for (int r = 0; r < 8; ++r) {
            float v0 = c0[r] + bb; v0 = v0 > 0.f ? v0 : 0.f;
            float v1 = c1[r] + bb; v1 = v1 > 0.f ? v1 : 0.f;
            h16[(r + g8row) * 128 + col]      = f2bf(v0);
            h16[(r + g8row + 16) * 128 + col] = f2bf(v1);
        }
    }
    __syncthreads();

    // ---- layer 3: 128 -> 128 (f32 out to LDS) ----
    {
        v8f c0 = {0.f,0.f,0.f,0.f,0.f,0.f,0.f,0.f};
        v8f c1 = {0.f,0.f,0.f,0.f,0.f,0.f,0.f,0.f};
        gemm_tile2<128>(H2s, Wt3, colBase, lane, c0, c1);
        float bb = b3[col];
#pragma unroll
        for (int r = 0; r < 8; ++r) {
            Os[(r + g8row) * 128 + col]      = c0[r] + bb;
            Os[(r + g8row + 16) * 128 + col] = c1[r] + bb;
        }
    }
    __syncthreads();

    // ---- LayerNorm over 128 cols per row (8 threads x 16 elems per row) ----
    const int rr = tid >> 3, ss = tid & 7;
    float sum = 0.f, sq = 0.f;
#pragma unroll
    for (int j = 0; j < 16; ++j) {
        float h = Os[rr * 128 + ss * 16 + j];
        sum += h; sq += h * h;
    }
    red[rr][ss][0] = sum; red[rr][ss][1] = sq;
    __syncthreads();
    if (ss == 0) {
        float S = 0.f, Q = 0.f;
#pragma unroll
        for (int i = 0; i < 8; ++i) { S += red[rr][i][0]; Q += red[rr][i][1]; }
        float mu  = S * (1.f / 128.f);
        float var = Q * (1.f / 128.f) - mu * mu;
        stats[rr][0] = mu;
        stats[rr][1] = rsqrtf(var + 1e-5f);
    }
    __syncthreads();

    const int erow = rowBase + rr;
    if (erow < nrows) {
        float mu = stats[rr][0], rs = stats[rr][1];
        int dnode = EDGE ? dstIdx[erow] : 0;
#pragma unroll
        for (int j = 0; j < 16; ++j) {
            int c = ss * 16 + j;
            float y = (Os[rr * 128 + c] - mu) * rs * gamma[c] + beta[c];
            y += resid[(size_t)erow * 128 + c];
            if (outBuf) outBuf[(size_t)erow * 128 + c] = y;
            if (EDGE)   atomicAdd(&aggBuf[(size_t)dnode * 128 + c], y);  // segment_sum fused
        }
    }
}

// Transpose K x N f32 weights -> N x K bf16 (WMMA-B-friendly layout).
__global__ void transpose_bf16_kernel(const float* __restrict__ W, u16* __restrict__ Wt,
                                      int K, int N) {
    int i = blockIdx.x * blockDim.x + threadIdx.x;
    if (i < K * N) {
        int k = i / N, n = i - k * N;
        Wt[n * K + k] = f2bf(W[i]);
    }
}

// n1[v] = (v < N1) ? n1[v] : n0[v]
__global__ void merge_kernel(float* __restrict__ n1, const float* __restrict__ n0,
                             int n1cnt, int total) {
    int i = blockIdx.x * blockDim.x + threadIdx.x;
    if (i < total && (i >> 7) >= n1cnt) n1[i] = n0[i];
}

// ---------------------------------------------------------------------------

extern "C" void kernel_launch(void* const* d_in, const int* in_sizes, int n_in,
                              void* d_out, int out_size, void* d_ws, size_t ws_size,
                              hipStream_t stream) {
    (void)in_sizes; (void)n_in; (void)out_size; (void)ws_size;
    const int Nn = 20000, Nl1 = 5000;
    const int Enb = 320000, Edn = 100000, Elr = 160000, Eup = 100000;

    const float* nodeF = (const float*)d_in[0];
    const float* efNb  = (const float*)d_in[1];
    const float* efDn  = (const float*)d_in[2];
    const float* efLr  = (const float*)d_in[3];
    const float* efUp  = (const float*)d_in[4];
    const int* srcNb = (const int*)d_in[5];
    const int* dstNb = (const int*)d_in[6];
    const int* srcDn = (const int*)d_in[7];
    const int* dstDn = (const int*)d_in[8];
    const int* srcLr = (const int*)d_in[9];
    const int* dstLr = (const int*)d_in[10];
    const int* srcUp = (const int*)d_in[11];
    const int* dstUp = (const int*)d_in[12];
    // d_in[13] = lvl1_mask, unused (mask == idx < N1)
    const int PBASE = 14;   // params: 10 MLPs x {W1,b1,W2,b2,W3,b3,g,beta}

    // ---- workspace layout ----
    char* ws = (char*)d_ws;
    const size_t WSLOT = (size_t)384 * 128 * 2 + 2 * (size_t)128 * 128 * 2; // 163840 B
    u16* wt[10][3];
    for (int m = 0; m < 10; ++m) {
        int kin = (m & 1) ? 256 : 384;     // even = edge MLP, odd = node MLP
        wt[m][0] = (u16*)(ws + (size_t)m * WSLOT);
        wt[m][1] = (u16*)(ws + (size_t)m * WSLOT + (size_t)kin * 128 * 2);
        wt[m][2] = (u16*)(ws + (size_t)m * WSLOT + (size_t)kin * 128 * 2 + (size_t)128 * 128 * 2);
    }
    size_t off = 10 * WSLOT;
    const size_t NODE_BYTES = (size_t)Nn * 128 * sizeof(float);
    float* agg = (float*)(ws + off); off += NODE_BYTES;
    float* n0  = (float*)(ws + off); off += NODE_BYTES;
    float* n1  = (float*)(ws + off); off += NODE_BYTES;
    float* enb = (float*)(ws + off); off += (size_t)Enb * 128 * sizeof(float);

    // ---- convert + transpose all weights to bf16 [col][k] ----
    for (int m = 0; m < 10; ++m) {
        int kin = (m & 1) ? 256 : 384;
        const float* W1 = (const float*)d_in[PBASE + 8 * m + 0];
        const float* W2 = (const float*)d_in[PBASE + 8 * m + 2];
        const float* W3 = (const float*)d_in[PBASE + 8 * m + 4];
        transpose_bf16_kernel<<<(kin * 128 + 255) / 256, 256, 0, stream>>>(W1, wt[m][0], kin, 128);
        transpose_bf16_kernel<<<(128 * 128 + 255) / 256, 256, 0, stream>>>(W2, wt[m][1], 128, 128);
        transpose_bf16_kernel<<<(128 * 128 + 255) / 256, 256, 0, stream>>>(W3, wt[m][2], 128, 128);
    }

    auto edge = [&](int m, const float* ef, const float* nf, const int* s, const int* d,
                    float* outB, int E) {
        (void)hipMemsetAsync(agg, 0, NODE_BYTES, stream);
        mgn_block_kernel<384, true><<<(E + 31) / 32, 256, 0, stream>>>(
            ef, nf, s, d, ef,
            wt[m][0], (const float*)d_in[PBASE + 8 * m + 1],
            wt[m][1], (const float*)d_in[PBASE + 8 * m + 3],
            wt[m][2], (const float*)d_in[PBASE + 8 * m + 5],
            (const float*)d_in[PBASE + 8 * m + 6], (const float*)d_in[PBASE + 8 * m + 7],
            outB, agg, E);
    };
    auto node = [&](int m, const float* nfin, const float* nfout, float* outB) {
        mgn_block_kernel<256, false><<<(Nn + 31) / 32, 256, 0, stream>>>(
            agg, nfin, nullptr, nullptr, nfout,
            wt[m][0], (const float*)d_in[PBASE + 8 * m + 1],
            wt[m][1], (const float*)d_in[PBASE + 8 * m + 3],
            wt[m][2], (const float*)d_in[PBASE + 8 * m + 5],
            (const float*)d_in[PBASE + 8 * m + 6], (const float*)d_in[PBASE + 8 * m + 7],
            outB, nullptr, Nn);
    };

    // ---- pipeline (matches reference dataflow) ----
    edge(0, efNb, nodeF, srcNb, dstNb, enb, Enb);      // e_nb (kept)
    node(1, nodeF, nodeF, n0);                          // n0 = in-layer node block
    edge(2, efDn, n0, srcDn, dstDn, nullptr, Edn);      // down edges (agg only)
    node(3, n0, n0, n1);                                // n1 = down node block
    edge(4, efLr, n1, srcLr, dstLr, nullptr, Elr);      // long-range edges
    node(5, n1, n1, n1);                                // n1 = lr node block (row-local, in-place OK)
    merge_kernel<<<(Nn * 128 + 255) / 256, 256, 0, stream>>>(n1, n0, Nl1, Nn * 128);
    edge(6, efUp, n1, srcUp, dstUp, nullptr, Eup);      // up edges
    node(7, n1, n0, n0);                                // n0 = up node block
    edge(8, enb, n0, srcNb, dstNb, nullptr, Enb);       // second neighbor edge block (resid = e_nb)
    node(9, n0, n0, (float*)d_out);                     // final node block -> output
}
